// UpSwin_89137751261668
// MI455X (gfx1250) — matrix operand order repaired and checked by reading, not compiled, not run on hardware
//
#include <hip/hip_runtime.h>
#include <math.h>

// ---------------------------------------------------------------------------
// Types for CDNA5 WMMA (wave32)
// ---------------------------------------------------------------------------
typedef __attribute__((ext_vector_type(16))) __bf16 v16bf;
typedef __attribute__((ext_vector_type(8)))  float  v8f;
typedef unsigned short bf16_t;

__device__ __forceinline__ bf16_t f2bf(float f) {
    unsigned u = __builtin_bit_cast(unsigned, f);
    u += 0x7FFFu + ((u >> 16) & 1u);   // round-to-nearest-even
    return (bf16_t)(u >> 16);
}

union FragU { uint4 q[2]; v16bf v; };

// ---------------------------------------------------------------------------
// Problem constants
// ---------------------------------------------------------------------------
#define DIM     256
#define HEADS   8
#define HEAD_D  256
#define INNER   2048     // HEADS*HEAD_D
#define QKV_LD  6144     // 3*INNER
#define WS7     7
#define NTOK    49       // tokens per window
#define HW      56
#define TOKENS  12544    // 4*56*56
#define ATT_SCALE 0.0625f // 256^-0.5

// GEMM epilogue modes
#define M_OUT_BF16 0
#define M_OUT_F32  1
#define M_RES_WIN  2
#define M_RES_LIN  3

// ---------------------------------------------------------------------------
// fp32 -> bf16 convert (row-major copy)
// ---------------------------------------------------------------------------
__global__ void convert_bf16_kernel(const float* __restrict__ in,
                                    bf16_t* __restrict__ out, int n) {
    int i = blockIdx.x * blockDim.x + threadIdx.x;
    if (i < n) out[i] = f2bf(in[i]);
}

// fp32 [K][N] -> bf16 [N][K] (one-time weight transpose+convert)
__global__ void transpose_bf16_kernel(const float* __restrict__ src,
                                      bf16_t* __restrict__ dst, int K, int N) {
    int idx = blockIdx.x * blockDim.x + threadIdx.x;
    if (idx < K * N) {
        int n = idx / K, k = idx - n * K;
        dst[idx] = f2bf(src[(size_t)k * N + n]);
    }
}

__global__ void copy_f32_kernel(const float* __restrict__ in,
                                float* __restrict__ out, int n) {
    int i = blockIdx.x * blockDim.x + threadIdx.x;
    if (i < n) out[i] = in[i];
}

// ---------------------------------------------------------------------------
// WMMA bf16 GEMM: C[M,N] = A[M,K](bf16) * Bt[N,K](bf16, pre-transposed) + bias
// 4 waves, 128x64 tile, K-step 32; each wave: 2 row strips x 4 col tiles.
// Global->LDS staging uses CDNA5 async-to-LDS ops (ASYNCcnt).
// ---------------------------------------------------------------------------
__global__ __launch_bounds__(128)
void gemm_wmma_kernel(const bf16_t* __restrict__ A,
                      const bf16_t* __restrict__ Bt,
                      const float*  __restrict__ bias,
                      bf16_t* __restrict__ outb,
                      float*  __restrict__ outf,
                      const float* __restrict__ resid,
                      int M, int N, int K,
                      int mode, int shift, int do_gelu) {
    __shared__ __align__(16) bf16_t sA[128 * 32];
    __shared__ __align__(16) bf16_t sB[64 * 32];

    const int tid  = threadIdx.x;
    const int lane = tid & 31;
    const int wave = tid >> 5;
    const int m0 = blockIdx.y * 128;
    const int n0 = blockIdx.x * 64;

    // staging assignments (per thread): A row = tid (clamped), B half-row
    int arow = m0 + tid; if (arow >= M) arow = M - 1;
    const bf16_t* aptr = A  + (size_t)arow * K;
    const bf16_t* bptr = Bt + (size_t)(n0 + (tid >> 1)) * K + (tid & 1) * 16;
    unsigned ldsA = (unsigned)(unsigned long long)(sA + tid * 32);
    unsigned ldsB = (unsigned)(unsigned long long)(sB + (tid >> 1) * 32 + (tid & 1) * 16);

    v8f acc[8] = {};

    for (int k0 = 0; k0 < K; k0 += 32) {
        unsigned long long ga = (unsigned long long)(aptr + k0);
        unsigned long long gb = (unsigned long long)(bptr + k0);
        // async global->LDS staging: A row (64B) + B half-row (32B)
        asm volatile(
            "global_load_async_to_lds_b128 %0, %1, off\n\t"
            "global_load_async_to_lds_b128 %0, %1, off offset:16\n\t"
            "global_load_async_to_lds_b128 %0, %1, off offset:32\n\t"
            "global_load_async_to_lds_b128 %0, %1, off offset:48\n\t"
            "global_load_async_to_lds_b128 %2, %3, off\n\t"
            "global_load_async_to_lds_b128 %2, %3, off offset:16\n\t"
            "s_wait_asynccnt 0x0"
            :: "v"(ldsA), "v"(ga), "v"(ldsB), "v"(gb)
            : "memory");
        __syncthreads();

        const int ml = lane & 15;
        const int kh = lane >> 4;
        FragU a0, a1;
        a0.q[0] = *(const uint4*)(sA + (wave * 16 + ml) * 32 + kh * 8);
        a0.q[1] = *(const uint4*)(sA + (wave * 16 + ml) * 32 + 16 + kh * 8);
        a1.q[0] = *(const uint4*)(sA + (64 + wave * 16 + ml) * 32 + kh * 8);
        a1.q[1] = *(const uint4*)(sA + (64 + wave * 16 + ml) * 32 + 16 + kh * 8);
#pragma unroll
        for (int t = 0; t < 4; t++) {
            FragU ub;
            ub.q[0] = *(const uint4*)(sB + (t * 16 + ml) * 32 + kh * 8);
            ub.q[1] = *(const uint4*)(sB + (t * 16 + ml) * 32 + 16 + kh * 8);
            acc[t] = __builtin_amdgcn_wmma_f32_16x16x32_bf16(
                false, a0.v, false, ub.v, (short)0, acc[t], false, false);
            acc[4 + t] = __builtin_amdgcn_wmma_f32_16x16x32_bf16(
                false, a1.v, false, ub.v, (short)0, acc[4 + t], false, false);
        }
        __syncthreads();
    }

    // --- epilogue
#pragma unroll
    for (int s = 0; s < 2; s++) {
#pragma unroll
        for (int t = 0; t < 4; t++) {
            int col = n0 + t * 16 + (lane & 15);
            float bv = bias ? bias[col] : 0.0f;
#pragma unroll
            for (int i = 0; i < 8; i++) {
                int row = m0 + s * 64 + wave * 16 + i + 8 * (lane >> 4);
                if (row >= M) continue;
                float v = acc[s * 4 + t][i] + bv;
                if (do_gelu) {
                    float x3 = v * v * v;
                    float tz = tanhf(0.7978845608028654f * (v + 0.044715f * x3));
                    v = 0.5f * v * (1.0f + tz);
                }
                if (mode == M_OUT_BF16) {
                    outb[(size_t)row * N + col] = f2bf(v);
                } else if (mode == M_OUT_F32) {
                    outf[(size_t)row * N + col] = v;
                } else if (mode == M_RES_LIN) {
                    size_t idx = (size_t)row * N + col;
                    outf[idx] = resid[idx] + v;
                } else { // M_RES_WIN
                    int win = row / NTOK, n = row - win * NTOK;
                    int b = win >> 6, rem = win & 63;
                    int hb = rem >> 3, wb = rem & 7;
                    int hr = hb * WS7 + n / WS7, wr = wb * WS7 + n % WS7;
                    int h = hr + shift; if (h >= HW) h -= HW;
                    int w = wr + shift; if (w >= HW) w -= HW;
                    size_t idx = ((size_t)((b * HW + h) * HW + w)) * DIM + col;
                    outf[idx] = resid[idx] + v;
                }
            }
        }
    }
}

// ---------------------------------------------------------------------------
// Pixel-shuffle + pe_norm LayerNorm: Ytmp[3136,1024] -> X[12544,256] (f32)
// ---------------------------------------------------------------------------
__global__ __launch_bounds__(128)
void shuffle_ln_kernel(const float* __restrict__ Ytmp,
                       const float* __restrict__ g,
                       const float* __restrict__ bta,
                       float* __restrict__ X) {
    int token = blockIdx.x * 4 + (threadIdx.x >> 5);
    int lane  = threadIdx.x & 31;
    int b = token / (HW * HW);
    int rr = token - b * HW * HW;
    int h = rr / HW, w = rr % HW;
    int hi = h >> 1, s1 = h & 1, wi = w >> 1, s2 = w & 1;
    const float* xr = Ytmp + (size_t)(b * 784 + hi * 28 + wi) * 1024
                           + (s1 * 2 + s2) * DIM;
    float vals[8];
    float s = 0.0f;
#pragma unroll
    for (int j = 0; j < 8; j++) { vals[j] = xr[lane + j * 32]; s += vals[j]; }
#pragma unroll
    for (int off = 16; off > 0; off >>= 1) s += __shfl_xor(s, off, 32);
    float mean = s * (1.0f / 256.0f);
    float vs = 0.0f;
#pragma unroll
    for (int j = 0; j < 8; j++) { float d = vals[j] - mean; vs += d * d; }
#pragma unroll
    for (int off = 16; off > 0; off >>= 1) vs += __shfl_xor(vs, off, 32);
    float rinv = rsqrtf(vs * (1.0f / 256.0f) + 1e-5f);
    float* out = X + (size_t)token * DIM;
#pragma unroll
    for (int j = 0; j < 8; j++) {
        int c = lane + j * 32;
        out[c] = (vals[j] - mean) * rinv * g[c] + bta[c];
    }
}

// ---------------------------------------------------------------------------
// LayerNorm (+optional roll & window reorder), f32 X -> bf16 Y
// ---------------------------------------------------------------------------
__global__ __launch_bounds__(128)
void ln_kernel(const float* __restrict__ X,
               const float* __restrict__ g,
               const float* __restrict__ bta,
               bf16_t* __restrict__ Y,
               int shift, int window_order) {
    int token = blockIdx.x * 4 + (threadIdx.x >> 5);
    int lane  = threadIdx.x & 31;
    int src;
    if (window_order) {
        int win = token / NTOK, n = token - win * NTOK;
        int b = win >> 6, rem = win & 63;
        int hb = rem >> 3, wb = rem & 7;
        int hr = hb * WS7 + n / WS7, wr = wb * WS7 + n % WS7;
        int h = hr + shift; if (h >= HW) h -= HW;   // y = roll(x, -shift)
        int w = wr + shift; if (w >= HW) w -= HW;
        src = (b * HW + h) * HW + w;
    } else {
        src = token;
    }
    const float* xr = X + (size_t)src * DIM;
    float vals[8];
    float s = 0.0f;
#pragma unroll
    for (int j = 0; j < 8; j++) { vals[j] = xr[lane + j * 32]; s += vals[j]; }
#pragma unroll
    for (int off = 16; off > 0; off >>= 1) s += __shfl_xor(s, off, 32);
    float mean = s * (1.0f / 256.0f);
    float vs = 0.0f;
#pragma unroll
    for (int j = 0; j < 8; j++) { float d = vals[j] - mean; vs += d * d; }
#pragma unroll
    for (int off = 16; off > 0; off >>= 1) vs += __shfl_xor(vs, off, 32);
    float rinv = rsqrtf(vs * (1.0f / 256.0f) + 1e-5f);
    bf16_t* out = Y + (size_t)token * DIM;
#pragma unroll
    for (int j = 0; j < 8; j++) {
        int c = lane + j * 32;
        out[c] = f2bf((vals[j] - mean) * rinv * g[c] + bta[c]);
    }
}

// ---------------------------------------------------------------------------
// Window attention, one block per (window, head). 4 waves.
// ---------------------------------------------------------------------------
__device__ __forceinline__ int region_h(int gp) { return gp < 49 ? 0 : (gp < 53 ? 1 : 2); }

__global__ __launch_bounds__(128)
void attn_kernel(const bf16_t* __restrict__ QKV,
                 const float*  __restrict__ rel_bias,  // [169,8] for block i
                 bf16_t* __restrict__ O,
                 int shifted) {
    const int win  = blockIdx.x >> 3;
    const int head = blockIdx.x & 7;
    const int tid  = threadIdx.x;
    const int lane = tid & 31;
    const int wave = tid >> 5;

    __shared__ __align__(16) float  sS[64 * 64];
    __shared__ __align__(16) bf16_t sP[64 * 64];
    __shared__ __align__(16) bf16_t sStage[2 * 64 * 32];
    bf16_t* sQ = sStage;
    bf16_t* sK = sStage + 64 * 32;
    bf16_t* sV = sStage;   // reused as [n_local][kk] 64x64 in PV phase

    const size_t base = (size_t)win * NTOK * QKV_LD + (size_t)head * HEAD_D;

    // ---------------- S = Q K^T ----------------
    {
        v8f acc[4] = {};
        for (int k0 = 0; k0 < HEAD_D; k0 += 32) {
            int row = tid >> 1, cs = (tid & 1) * 16;
            uint4 z = make_uint4(0, 0, 0, 0);
            uint4 q0 = z, q1 = z, kk0 = z, kk1 = z;
            if (row < NTOK) {
                const uint4* sq = (const uint4*)(QKV + base + (size_t)row * QKV_LD + k0 + cs);
                q0 = sq[0]; q1 = sq[1];
                const uint4* sk = (const uint4*)(QKV + base + INNER + (size_t)row * QKV_LD + k0 + cs);
                kk0 = sk[0]; kk1 = sk[1];
            }
            uint4* dq = (uint4*)(sQ + row * 32 + cs); dq[0] = q0; dq[1] = q1;
            uint4* dk = (uint4*)(sK + row * 32 + cs); dk[0] = kk0; dk[1] = kk1;
            __syncthreads();

            const int mrow = wave * 16 + (lane & 15);
            const int kh   = lane >> 4;
            FragU ua;
            ua.q[0] = *(const uint4*)(sQ + mrow * 32 + kh * 8);
            ua.q[1] = *(const uint4*)(sQ + mrow * 32 + 16 + kh * 8);
#pragma unroll
            for (int t = 0; t < 4; t++) {
                int nn = t * 16 + (lane & 15);
                FragU ub;
                ub.q[0] = *(const uint4*)(sK + nn * 32 + kh * 8);
                ub.q[1] = *(const uint4*)(sK + nn * 32 + 16 + kh * 8);
                acc[t] = __builtin_amdgcn_wmma_f32_16x16x32_bf16(
                    false, ua.v, false, ub.v, (short)0, acc[t], false, false);
            }
            __syncthreads();
        }
#pragma unroll
        for (int t = 0; t < 4; t++) {
            int nn = t * 16 + (lane & 15);
#pragma unroll
            for (int i = 0; i < 8; i++) {
                int mm = wave * 16 + i + 8 * (lane >> 4);
                sS[mm * 64 + nn] = acc[t][i] * ATT_SCALE;
            }
        }
    }
    __syncthreads();

    // ---------------- softmax rows ----------------
    if (tid < 64) {
        int n = tid;
        if (n < NTOK) {
            int rem = win & 63;
            int hb = rem >> 3, wb = rem & 7;
            int hn = n / WS7, wn = n % WS7;
            int rhn = region_h(hb * WS7 + hn), rwn = region_h(wb * WS7 + wn);
            float mx = -1e30f;
            for (int j = 0; j < NTOK; j++) {
                int hm = j / WS7, wm = j % WS7;
                float v = sS[n * 64 + j]
                        + rel_bias[((hn - hm + 6) * 13 + (wn - wm + 6)) * HEADS + head];
                if (shifted) {
                    if (rhn != region_h(hb * WS7 + hm) ||
                        rwn != region_h(wb * WS7 + wm)) v -= 100.0f;
                }
                sS[n * 64 + j] = v;
                mx = fmaxf(mx, v);
            }
            float sum = 0.0f;
            for (int j = 0; j < NTOK; j++) {
                float e = __expf(sS[n * 64 + j] - mx);
                sS[n * 64 + j] = e;
                sum += e;
            }
            float inv = 1.0f / sum;
            for (int j = 0; j < 64; j++)
                sP[n * 64 + j] = (j < NTOK) ? f2bf(sS[n * 64 + j] * inv) : (bf16_t)0;
        } else {
            for (int j = 0; j < 64; j++) sP[n * 64 + j] = (bf16_t)0;
        }
    }
    __syncthreads();

    // ---------------- O = P V, 4 column chunks of 64 ----------------
    for (int cc = 0; cc < 4; cc++) {
        {
            int kk = tid >> 1, hf = (tid & 1) * 32;
            if (kk < NTOK) {
                const bf16_t* vr = QKV + base + 2 * INNER + (size_t)kk * QKV_LD + cc * 64 + hf;
#pragma unroll
                for (int j = 0; j < 32; j++) sV[(hf + j) * 64 + kk] = vr[j];
            } else {
#pragma unroll
                for (int j = 0; j < 32; j++) sV[(hf + j) * 64 + kk] = (bf16_t)0;
            }
        }
        __syncthreads();

        v8f acc[4] = {};
        const int mrow = wave * 16 + (lane & 15);
        const int kh   = lane >> 4;
#pragma unroll
        for (int kstep = 0; kstep < 2; kstep++) {
            int kk0 = kstep * 32;
            FragU ua;
            ua.q[0] = *(const uint4*)(sP + mrow * 64 + kk0 + kh * 8);
            ua.q[1] = *(const uint4*)(sP + mrow * 64 + kk0 + 16 + kh * 8);
#pragma unroll
            for (int t = 0; t < 4; t++) {
                int nn = t * 16 + (lane & 15);
                FragU ub;
                ub.q[0] = *(const uint4*)(sV + nn * 64 + kk0 + kh * 8);
                ub.q[1] = *(const uint4*)(sV + nn * 64 + kk0 + 16 + kh * 8);
                acc[t] = __builtin_amdgcn_wmma_f32_16x16x32_bf16(
                    false, ua.v, false, ub.v, (short)0, acc[t], false, false);
            }
        }
#pragma unroll
        for (int t = 0; t < 4; t++) {
            int nn = t * 16 + (lane & 15);
#pragma unroll
            for (int i = 0; i < 8; i++) {
                int mm = wave * 16 + i + 8 * (lane >> 4);
                if (mm < NTOK) {
                    O[((size_t)win * NTOK + mm) * INNER + head * HEAD_D + cc * 64 + nn]
                        = f2bf(acc[t][i]);
                }
            }
        }
        __syncthreads();
    }
}

// ---------------------------------------------------------------------------
// Host-side orchestration
// ---------------------------------------------------------------------------
extern "C" void kernel_launch(void* const* d_in, const int* in_sizes, int n_in,
                              void* d_out, int out_size, void* d_ws, size_t ws_size,
                              hipStream_t stream) {
    const float* x        = (const float*)d_in[0];
    const float* expand_w = (const float*)d_in[1];
    const float* expand_b = (const float*)d_in[2];
    const float* pe_g     = (const float*)d_in[3];
    const float* pe_b     = (const float*)d_in[4];
    const float* n1_g     = (const float*)d_in[5];
    const float* n1_b     = (const float*)d_in[6];
    const float* qkv_w    = (const float*)d_in[7];
    const float* qkv_b    = (const float*)d_in[8];
    const float* proj_w   = (const float*)d_in[9];
    const float* proj_b   = (const float*)d_in[10];
    const float* rel_bias = (const float*)d_in[11];
    const float* n2_g     = (const float*)d_in[12];
    const float* n2_b     = (const float*)d_in[13];
    const float* mlp_w1   = (const float*)d_in[14];
    const float* mlp_b1   = (const float*)d_in[15];
    const float* mlp_w2   = (const float*)d_in[16];
    const float* mlp_b2   = (const float*)d_in[17];
    float* out = (float*)d_out;

    // workspace carve (all 256B aligned)
    char* w = (char*)d_ws;
    float*  X    = (float*)(w);                       // 12544x256 f32
    float*  Ytmp = (float*)(w + 12845056);            // 3136x1024 f32
    bf16_t* Ybf  = (bf16_t*)(w + 25690112);           // 12544x256 bf16
    bf16_t* QKV  = (bf16_t*)(w + 32112640);           // 12544x6144 bf16
    bf16_t* Obuf = (bf16_t*)(w + 186253312);          // 12544x2048 bf16
    bf16_t* Hbuf = (bf16_t*)(w + 237633536);          // 12544x1024 bf16
    bf16_t* Xbf  = (bf16_t*)(w + 263323648);          // 3136x512 bf16
    // pre-transposed bf16 weights [N][K]
    bf16_t* exp_wt  = (bf16_t*)(w + 266534912);       // 1024x512
    bf16_t* qkv_wt  = (bf16_t*)(w + 267583488);       // 2x 6144x256
    bf16_t* proj_wt = (bf16_t*)(w + 273874944);       // 2x 256x2048
    bf16_t* mlp1_wt = (bf16_t*)(w + 275972096);       // 2x 1024x256
    bf16_t* mlp2_wt = (bf16_t*)(w + 277020672);       // 2x 256x1024

    // 0) one-time weight transpose+convert (fp32 [K][N] -> bf16 [N][K])
    transpose_bf16_kernel<<<(512 * 1024 + 255) / 256, 256, 0, stream>>>(
        expand_w, exp_wt, 512, 1024);
    for (int i = 0; i < 2; i++) {
        transpose_bf16_kernel<<<(256 * 6144 + 255) / 256, 256, 0, stream>>>(
            qkv_w + (size_t)i * 256 * 6144, qkv_wt + (size_t)i * 6144 * 256, 256, 6144);
        transpose_bf16_kernel<<<(2048 * 256 + 255) / 256, 256, 0, stream>>>(
            proj_w + (size_t)i * 2048 * 256, proj_wt + (size_t)i * 256 * 2048, 2048, 256);
        transpose_bf16_kernel<<<(256 * 1024 + 255) / 256, 256, 0, stream>>>(
            mlp_w1 + (size_t)i * 256 * 1024, mlp1_wt + (size_t)i * 1024 * 256, 256, 1024);
        transpose_bf16_kernel<<<(1024 * 256 + 255) / 256, 256, 0, stream>>>(
            mlp_w2 + (size_t)i * 1024 * 256, mlp2_wt + (size_t)i * 256 * 1024, 1024, 256);
    }

    // 1) x -> bf16
    {
        int n = 3136 * 512;
        convert_bf16_kernel<<<(n + 255) / 256, 256, 0, stream>>>(x, Xbf, n);
    }
    // 2) expand GEMM: [3136,512] @ [512,1024] + b -> Ytmp f32
    gemm_wmma_kernel<<<dim3(1024 / 64, (3136 + 127) / 128), 128, 0, stream>>>(
        Xbf, exp_wt, expand_b, nullptr, Ytmp, nullptr,
        3136, 1024, 512, M_OUT_F32, 0, 0);
    // 3) pixel shuffle + pe_norm -> X
    shuffle_ln_kernel<<<TOKENS / 4, 128, 0, stream>>>(Ytmp, pe_g, pe_b, X);

    for (int i = 0; i < 2; i++) {
        const int shift = (i == 1) ? (WS7 / 2) : 0;
        // 4) LN1 + roll + window order -> Ybf
        ln_kernel<<<TOKENS / 4, 128, 0, stream>>>(
            X, n1_g + i * DIM, n1_b + i * DIM, Ybf, shift, 1);
        // 5) QKV GEMM -> QKV bf16
        gemm_wmma_kernel<<<dim3(QKV_LD / 64, TOKENS / 128), 128, 0, stream>>>(
            Ybf, qkv_wt + (size_t)i * QKV_LD * DIM, qkv_b + (size_t)i * QKV_LD,
            QKV, nullptr, nullptr, TOKENS, QKV_LD, DIM, M_OUT_BF16, 0, 0);
        // 6) attention: 256 windows x 8 heads
        attn_kernel<<<256 * HEADS, 128, 0, stream>>>(
            QKV, rel_bias + (size_t)i * 169 * HEADS, Obuf, i == 1 ? 1 : 0);
        // 7) proj GEMM + window residual -> X (in place)
        gemm_wmma_kernel<<<dim3(DIM / 64, TOKENS / 128), 128, 0, stream>>>(
            Obuf, proj_wt + (size_t)i * DIM * INNER, proj_b + (size_t)i * DIM,
            nullptr, X, X, TOKENS, DIM, INNER, M_RES_WIN, shift, 0);
        // 8) LN2 (linear order) -> Ybf
        ln_kernel<<<TOKENS / 4, 128, 0, stream>>>(
            X, n2_g + i * DIM, n2_b + i * DIM, Ybf, 0, 0);
        // 9) MLP1 + GELU -> Hbuf bf16
        gemm_wmma_kernel<<<dim3(1024 / 64, TOKENS / 128), 128, 0, stream>>>(
            Ybf, mlp1_wt + (size_t)i * 1024 * DIM, mlp_b1 + (size_t)i * 1024,
            Hbuf, nullptr, nullptr, TOKENS, 1024, DIM, M_OUT_BF16, 0, 1);
        // 10) MLP2 + linear residual -> X
        gemm_wmma_kernel<<<dim3(DIM / 64, TOKENS / 128), 128, 0, stream>>>(
            Hbuf, mlp2_wt + (size_t)i * DIM * 1024, mlp_b2 + (size_t)i * DIM,
            nullptr, X, X, TOKENS, DIM, 1024, M_RES_LIN, 0, 0);
    }

    // 11) copy X -> d_out
    {
        int n = TOKENS * DIM;
        copy_f32_kernel<<<(n + 255) / 256, 256, 0, stream>>>(X, out, n);
    }
}
